// CausalSelfAttention_87840671138135
// MI455X (gfx1250) — matrix operand order repaired and checked by reading, not compiled, not run on hardware
//
#include <hip/hip_runtime.h>
#include <stdint.h>

typedef __bf16 bf16;
typedef __attribute__((ext_vector_type(16))) __bf16 v16bf;
typedef __attribute__((ext_vector_type(8)))  __bf16 v8bf;
typedef __attribute__((ext_vector_type(8)))  float  v8f;
typedef __attribute__((ext_vector_type(4)))  int    v4i;

#define SEQ     4096
#define DMODEL  1024
#define NHEAD   16
#define DHEAD   64

union V16 { v16bf v; v8bf h[2]; };

// ---------- async global->LDS (CDNA5), with safe fallback ----------

#if __has_builtin(__builtin_amdgcn_global_load_async_to_lds_b128)
#define HAS_ASYNC_LDS 1
#else
#define HAS_ASYNC_LDS 0
#endif

#define AS1 __attribute__((address_space(1)))
#define AS3 __attribute__((address_space(3)))

__device__ inline void lds_copy16(bf16* l, const bf16* g) {
#if HAS_ASYNC_LDS
  __builtin_amdgcn_global_load_async_to_lds_b128(
      (AS1 v4i*)(uintptr_t)g, (AS3 v4i*)(uint32_t)(uintptr_t)l, 0, 0);
#else
  *(v8bf*)l = *(const v8bf*)g;
#endif
}

__device__ inline void wait_async_all() {
#if HAS_ASYNC_LDS
#if __has_builtin(__builtin_amdgcn_s_wait_asynccnt)
  __builtin_amdgcn_s_wait_asynccnt(0);
#else
  asm volatile("s_wait_asynccnt 0x0" ::: "memory");
#endif
#endif
}

// ---------- WMMA helpers (layouts per CDNA5 ISA 7.12.2, wave32) ----------

__device__ inline v8f wmma_bf16(v16bf a, v16bf b, v8f c) {
  return __builtin_amdgcn_wmma_f32_16x16x32_bf16(
      false, a, false, b, (short)0, c, false, false);
}

// A fragment: row-major source, row stride `ld` halves. Lane L holds row L&15,
// halves[0..7]=A[r][kh..), halves[8..15]=A[r][16+kh..), kh=(L>>4)*8.
__device__ inline v16bf load_a_frag(const bf16* p, int ld) {
  const int lane = threadIdx.x & 31;
  const int r = lane & 15, kh = (lane >> 4) * 8;
  const bf16* q = p + r * ld + kh;
  V16 o;
  o.h[0] = *(const v8bf*)(q);
  o.h[1] = *(const v8bf*)(q + 16);
  return o.v;
}

// B fragment from transposed storage T[n][k]: B[k][n] = T[n][k].
// Lane L holds column n=L&15, halves[0..7]=k0.., halves[8..15]=16+k0.., k0=(L>>4)*8.
__device__ inline v16bf load_bt_frag(const bf16* p, int ld) {
  const int lane = threadIdx.x & 31;
  const int n = lane & 15, k0 = (lane >> 4) * 8;
  const bf16* q = p + n * ld + k0;
  V16 o;
  o.h[0] = *(const v8bf*)(q);
  o.h[1] = *(const v8bf*)(q + 16);
  return o.v;
}

// ---------- fp32 -> bf16 conversion ----------

__global__ void cvt_f32_bf16(const float* __restrict__ in,
                             bf16* __restrict__ out, int n) {
  int i = blockIdx.x * blockDim.x + threadIdx.x;
  if (i < n) out[i] = (bf16)in[i];
}

// vt[h*64+d][t] = qkv[t][2048 + h*64 + d]
__global__ void transpose_v(const bf16* __restrict__ qkv,
                            bf16* __restrict__ vt) {
  int idx = blockIdx.x * blockDim.x + threadIdx.x;   // t*1024 + c
  int t = idx >> 10, c = idx & 1023;
  vt[c * SEQ + t] = qkv[t * (3 * DMODEL) + 2 * DMODEL + c];
}

// ---------- bf16 WMMA GEMM: C[M,N] = A[M,K] * B[K,N] ----------
// block tile 128x128, 8 waves (4 in M x 2 in N), each wave 32x64 = 2x4 tiles.
// Double-buffered LDS with async global->LDS staging.

template <bool C_F32>
__global__ __launch_bounds__(256) void gemm_bf16(
    const bf16* __restrict__ A, const bf16* __restrict__ B,
    void* __restrict__ C, int M, int N, int K) {
  __shared__ bf16 ldsA[2][128 * 40];     // stride 40 halves (80B, 16B-aligned)
  __shared__ bf16 ldsBt[2][128 * 40];    // transposed B tile [n][k]
  const int tid = threadIdx.x;
  const int wave = tid >> 5, lane = tid & 31;
  const int wm = wave >> 1, wn = wave & 1;
  const int mb = blockIdx.y, nb = blockIdx.x;
  const int hi = lane >> 4, nlo = lane & 15;

  v8f acc[2][4] = {};

  auto stage = [&](int buf, int k0) {
    { // A tile 128x32 row-major: straight copy -> async
      int r = tid >> 1, ch = (tid & 1) * 16;
      const bf16* g = A + (size_t)(mb * 128 + r) * K + k0 + ch;
      bf16* l = &ldsA[buf][r * 40 + ch];
      lds_copy16(l, g);
      lds_copy16(l + 8, g + 8);
    }
    { // B tile 32x128 -> transpose scatter into ldsBt[n][k]
      int kk = tid >> 3, nch = (tid & 7) * 16;
      const bf16* g = B + (size_t)(k0 + kk) * N + nb * 128 + nch;
      v8bf b0 = *(const v8bf*)g;
      v8bf b1 = *(const v8bf*)(g + 8);
#pragma unroll
      for (int i = 0; i < 8; ++i) ldsBt[buf][(nch + i) * 40 + kk] = b0[i];
#pragma unroll
      for (int i = 0; i < 8; ++i) ldsBt[buf][(nch + 8 + i) * 40 + kk] = b1[i];
    }
  };

  const int niter = K / 32;
  stage(0, 0);
  for (int i = 0; i < niter; ++i) {
    wait_async_all();
    __syncthreads();
    if (i + 1 < niter) stage((i + 1) & 1, (i + 1) * 32);

    const int cur = i & 1;
    v16bf af[2], bf[4];
#pragma unroll
    for (int mt = 0; mt < 2; ++mt)
      af[mt] = load_a_frag(&ldsA[cur][(wm * 32 + mt * 16) * 40], 40);
#pragma unroll
    for (int nt = 0; nt < 4; ++nt)
      bf[nt] = load_bt_frag(&ldsBt[cur][(wn * 64 + nt * 16) * 40], 40);
#pragma unroll
    for (int mt = 0; mt < 2; ++mt)
#pragma unroll
      for (int nt = 0; nt < 4; ++nt)
        acc[mt][nt] = wmma_bf16(af[mt], bf[nt], acc[mt][nt]);
  }

  // epilogue: one 64-bit base per (mt), 32-bit offsets inside
  const int rbase = mb * 128 + wm * 32 + 8 * hi;
  const int cbase = nb * 128 + wn * 64 + nlo;
  if (C_F32) {
    float* Cf = (float*)C;
#pragma unroll
    for (int mt = 0; mt < 2; ++mt) {
      float* p = Cf + (size_t)(rbase + mt * 16) * N + cbase;
#pragma unroll
      for (int nt = 0; nt < 4; ++nt)
#pragma unroll
        for (int j = 0; j < 8; ++j)
          p[j * N + nt * 16] = acc[mt][nt][j];
    }
  } else {
    bf16* Cb = (bf16*)C;
#pragma unroll
    for (int mt = 0; mt < 2; ++mt) {
      bf16* p = Cb + (size_t)(rbase + mt * 16) * N + cbase;
#pragma unroll
      for (int nt = 0; nt < 4; ++nt)
#pragma unroll
        for (int j = 0; j < 8; ++j)
          p[j * N + nt * 16] = (bf16)acc[mt][nt][j];
    }
  }
}

// ---------- flash attention (bf16 WMMA, online softmax) ----------
// grid: (SEQ/128, NHEAD); 8 waves, wave w owns query rows [qb+16w, +16).
// K/Vt key-blocks of 32 double-buffered in LDS with async staging.

__global__ __launch_bounds__(256) void attn_kernel(
    const bf16* __restrict__ qkv, const bf16* __restrict__ vt,
    bf16* __restrict__ y) {
  __shared__ bf16 ldsK[2][32 * 72];     // [kk][d], stride 72 halves (144B)
  __shared__ bf16 ldsVt[2][64 * 40];    // [d][kk], stride 40 halves (80B)
  __shared__ bf16 ldsP[8][16 * 40];     // per-wave P tile 16x32

  const int tid = threadIdx.x;
  const int wave = tid >> 5, lane = tid & 31;
  const int h = blockIdx.y;
  const int qb = blockIdx.x * 128;
  const int qw = qb + wave * 16;
  const int hi = lane >> 4, nlo = lane & 15;

  // Q fragments for contraction d=[0,32) and [32,64)
  v16bf qf[2];
  {
    const bf16* base = qkv + (size_t)(qw + nlo) * (3 * DMODEL) + h * DHEAD;
    const int kh = hi * 8;
#pragma unroll
    for (int c = 0; c < 2; ++c) {
      V16 t;
      t.h[0] = *(const v8bf*)(base + c * 32 + kh);
      t.h[1] = *(const v8bf*)(base + c * 32 + kh + 16);
      qf[c] = t.v;
    }
  }

  v8f o[4] = {};
  float mrow[8], lrow[8];
#pragma unroll
  for (int j = 0; j < 8; ++j) { mrow[j] = -__builtin_inff(); lrow[j] = 0.f; }

  auto stage = [&](int buf, int kb) {
    { int kk = tid >> 3, dch = (tid & 7) * 8;
      lds_copy16(&ldsK[buf][kk * 72 + dch],
                 qkv + (size_t)(kb + kk) * (3 * DMODEL) + DMODEL + h * DHEAD + dch); }
    { int d = tid >> 2, kch = (tid & 3) * 8;
      lds_copy16(&ldsVt[buf][d * 40 + kch],
                 vt + (size_t)(h * DHEAD + d) * SEQ + kb + kch); }
  };

  const int nkb = (qb + 128) / 32;
  stage(0, 0);
  for (int i = 0; i < nkb; ++i) {
    wait_async_all();
    __syncthreads();
    if (i + 1 < nkb) stage((i + 1) & 1, (i + 1) * 32);

    const int kb = i * 32;
    const int cur = i & 1;
    if (kb <= qw + 15) {            // wave-uniform causal skip
      // S = Q K^T : 2 N-tiles x 2 K-subtiles
      v8f s[2] = {};
#pragma unroll
      for (int nt = 0; nt < 2; ++nt)
#pragma unroll
        for (int c = 0; c < 2; ++c) {
          v16bf kf = load_bt_frag(&ldsK[cur][nt * 16 * 72 + c * 32], 72);
          s[nt] = wmma_bf16(qf[c], kf, s[nt]);
        }
      // scale + causal mask (acc layout: elem(M=j+8*hi, N=nlo))
      const float scale = 0.125f;   // 1/sqrt(64)
#pragma unroll
      for (int nt = 0; nt < 2; ++nt)
#pragma unroll
        for (int j = 0; j < 8; ++j) {
          int m = qw + j + 8 * hi;
          int k = kb + nt * 16 + nlo;
          float v = s[nt][j] * scale;
          s[nt][j] = (k <= m) ? v : -__builtin_inff();
        }
      // online softmax; rows live in 16-lane groups -> xor<=8 reductions
#pragma unroll
      for (int j = 0; j < 8; ++j) {
        float rmax = fmaxf(s[0][j], s[1][j]);
#pragma unroll
        for (int off = 1; off <= 8; off <<= 1)
          rmax = fmaxf(rmax, __shfl_xor(rmax, off, 32));
        float mnew = fmaxf(mrow[j], rmax);
        float corr = __expf(mrow[j] - mnew);
        float p0 = __expf(s[0][j] - mnew);
        float p1 = __expf(s[1][j] - mnew);
        s[0][j] = p0; s[1][j] = p1;
        float rsum = p0 + p1;
#pragma unroll
        for (int off = 1; off <= 8; off <<= 1)
          rsum += __shfl_xor(rsum, off, 32);
        lrow[j] = lrow[j] * corr + rsum;
        mrow[j] = mnew;
#pragma unroll
        for (int nt = 0; nt < 4; ++nt) o[nt][j] *= corr;
      }
      // P (acc layout) -> per-wave LDS 16x32 row-major, reload as A fragment
      bf16* pw = &ldsP[wave][0];
#pragma unroll
      for (int nt = 0; nt < 2; ++nt)
#pragma unroll
        for (int j = 0; j < 8; ++j)
          pw[(j + 8 * hi) * 40 + nt * 16 + nlo] = (bf16)s[nt][j];
      v16bf pf = load_a_frag(pw, 40);
      // O += P(16x32) * V(32x64): 4 N-tiles
#pragma unroll
      for (int nt = 0; nt < 4; ++nt) {
        v16bf vf = load_bt_frag(&ldsVt[cur][nt * 16 * 40], 40);
        o[nt] = wmma_bf16(pf, vf, o[nt]);
      }
    }
  }

  // epilogue: y[t][h*64 + d] = O / l  (bf16), one 64-bit base only
  bf16* p = y + (size_t)(qw + 8 * hi) * DMODEL + h * DHEAD + nlo;
#pragma unroll
  for (int j = 0; j < 8; ++j) {
    float inv = 1.0f / lrow[j];
#pragma unroll
    for (int nt = 0; nt < 4; ++nt)
      p[j * DMODEL + nt * 16] = (bf16)(o[nt][j] * inv);
  }
}

// ---------- launch ----------

extern "C" void kernel_launch(void* const* d_in, const int* in_sizes, int n_in,
                              void* d_out, int out_size, void* d_ws, size_t ws_size,
                              hipStream_t stream) {
  const float* x      = (const float*)d_in[0];
  const float* w_qkv  = (const float*)d_in[1];
  const float* w_proj = (const float*)d_in[2];
  float* out = (float*)d_out;

  char* ws = (char*)d_ws;
  const size_t OFF_XB    = 0;
  const size_t OFF_WQKV  = OFF_XB    + (size_t)SEQ * DMODEL * 2;
  const size_t OFF_WPROJ = OFF_WQKV  + (size_t)DMODEL * 3 * DMODEL * 2;
  const size_t OFF_QKV   = OFF_WPROJ + (size_t)DMODEL * DMODEL * 2;
  const size_t OFF_VT    = OFF_QKV   + (size_t)SEQ * 3 * DMODEL * 2;
  const size_t OFF_Y     = OFF_VT    + (size_t)SEQ * DMODEL * 2;

  bf16* xb     = (bf16*)(ws + OFF_XB);
  bf16* wqkvb  = (bf16*)(ws + OFF_WQKV);
  bf16* wprojb = (bf16*)(ws + OFF_WPROJ);
  bf16* qkvb   = (bf16*)(ws + OFF_QKV);
  bf16* vtb    = (bf16*)(ws + OFF_VT);
  bf16* yb     = (bf16*)(ws + OFF_Y);

  const int nx = SEQ * DMODEL, nwq = DMODEL * 3 * DMODEL, nwp = DMODEL * DMODEL;
  cvt_f32_bf16<<<(nx  + 255) / 256, 256, 0, stream>>>(x, xb, nx);
  cvt_f32_bf16<<<(nwq + 255) / 256, 256, 0, stream>>>(w_qkv, wqkvb, nwq);
  cvt_f32_bf16<<<(nwp + 255) / 256, 256, 0, stream>>>(w_proj, wprojb, nwp);

  // qkv = x @ w_qkv  (bf16 out)
  gemm_bf16<false><<<dim3(3 * DMODEL / 128, SEQ / 128), 256, 0, stream>>>(
      xb, wqkvb, qkvb, SEQ, 3 * DMODEL, DMODEL);

  transpose_v<<<(SEQ * DMODEL) / 256, 256, 0, stream>>>(qkvb, vtb);

  attn_kernel<<<dim3(SEQ / 128, NHEAD), 256, 0, stream>>>(qkvb, vtb, yb);

  // out = y @ w_proj  (fp32 out)
  gemm_bf16<true><<<dim3(DMODEL / 128, SEQ / 128), 256, 0, stream>>>(
      yb, wprojb, out, SEQ, DMODEL, DMODEL);
}
